// TGAT_13202729467944
// MI455X (gfx1250) — matrix operand order repaired and checked by reading, not compiled, not run on hardware
//
#include <hip/hip_runtime.h>

// TGAT fused attention for gfx1250 (MI455X), f16 WMMA path.
// Pipeline: [swizzle weights] -> [Q GEMM] -> [fused KV GEMM + attention] -> [Wr GEMM + LN]

typedef __attribute__((ext_vector_type(16))) _Float16 v16h;
typedef __attribute__((ext_vector_type(8)))  float    v8f;

#define BB      20000
#define NN      20
#define NODE_D  172
#define EDGE_D  172
#define TIME_D  100
#define QD      272      // H*HEAD_D
#define KD      444
#define HEADD   136
#define NH      2
#define NTILES  17       // QD / 16
#define KT_KV   14       // ceil(444/32)
#define KT_Q    9        // ceil(272/32) -> pad to 288
#define KVP     448      // padded kv row (halfs)
#define QP      288      // padded q row (halfs)

__device__ __constant__ const float kScale = 0.08574929257125442f; // 136^-0.5
__device__ __constant__ const float kEps   = 1e-5f;

// ---- workspace layout (bytes) ----
constexpr size_t SZ_BQ  = (size_t)NTILES * KT_Q  * 512 * 2;   // 156,672
constexpr size_t SZ_BK  = (size_t)NTILES * KT_KV * 512 * 2;   // 243,712
constexpr size_t OFF_BQ = 0;
constexpr size_t OFF_BR = OFF_BQ + SZ_BQ;
constexpr size_t OFF_BK = OFF_BR + SZ_BQ;
constexpr size_t OFF_BV = OFF_BK + SZ_BK;
constexpr size_t OFF_Q  = OFF_BV + SZ_BK;                     // B*QD f32
constexpr size_t OFF_AO = OFF_Q + (size_t)BB * QD * 4;        // B*QD f32

// ---------------------------------------------------------------------------
// Kernel 0: pack W [272, IN] (row-major, out x in) into WMMA B-operand tiles.
// Tile (nt, kt) = 512 f16; lane l holds 16 contiguous halfs:
//   col = nt*16 + (l&15);  k = kt*32 + (l>>4)*16 + e   (zero-padded past IN)
// ---------------------------------------------------------------------------
__global__ __launch_bounds__(256)
void swizzle_w(const float* __restrict__ W, _Float16* __restrict__ dst,
               int IN, int KT) {
  int t = blockIdx.x * 256 + threadIdx.x;
  int total = NTILES * KT * 512;
  if (t >= total) return;
  int e    = t & 15;
  int lane = (t >> 4) & 31;
  int tile = t >> 9;
  int kt = tile % KT;
  int nt = tile / KT;
  int col = nt * 16 + (lane & 15);
  int k   = kt * 32 + (lane >> 4) * 16 + e;
  float v = (k < IN) ? W[(size_t)col * IN + k] : 0.0f;
  dst[t] = (_Float16)v;
}

// A-operand fetch helper pattern (contiguous k, same order as the B tiles):
// lane l: row = l&15; halfs 0..15 = K[base .. base+15]
// where base = kt*32 + (l>>4)*16.

// ---------------------------------------------------------------------------
// Kernel 1: Q = q_in @ Wq^T  (16 roots per block)
// ---------------------------------------------------------------------------
__global__ __launch_bounds__(256)
void q_gemm(const float* __restrict__ nodef, const float* __restrict__ timef,
            const _Float16* __restrict__ Bw, float* __restrict__ Qout) {
  __shared__ _Float16 A[16 * QP];
  int tid = threadIdx.x;
  int blk = blockIdx.x;
  for (int i = tid; i < 16 * QP; i += 256) {
    int r = i / QP, c = i % QP;
    size_t root = (size_t)blk * 16 + r;
    float v = 0.0f;
    if (c < NODE_D)      v = nodef[root * NODE_D + c];
    else if (c < QD)     v = timef[root * TIME_D + (c - NODE_D)];
    A[i] = (_Float16)v;
  }
  __syncthreads();

  int lane = tid & 31, wv = tid >> 5;
  int arow = lane & 15;
  int aoff = (lane >> 4) * 8;
  for (int nt = wv; nt < NTILES; nt += 8) {
    v8f acc = {};
    for (int kt = 0; kt < KT_Q; kt++) {
      union { uint4 u4[2]; v16h h; } a, b;
      const uint4* pa = (const uint4*)&A[arow * QP + kt * 32 + (lane >> 4) * 16];
      a.u4[0] = pa[0]; a.u4[1] = pa[1];
      const uint4* pb = (const uint4*)&Bw[((size_t)nt * KT_Q + kt) * 512 + lane * 16];
      b.u4[0] = pb[0]; b.u4[1] = pb[1];
      acc = __builtin_amdgcn_wmma_f32_16x16x32_f16(false, a.h, false, b.h,
                                                   (short)0, acc, false, false);
    }
    int col = nt * 16 + arow;
    for (int r = 0; r < 8; r++) {
      size_t root = (size_t)blk * 16 + aoff + r;
      Qout[root * QD + col] = acc[r];
    }
  }
}

// ---------------------------------------------------------------------------
// Kernel 2: fused K/V GEMMs + masked softmax attention (4 roots per block).
// 80 kv rows = 5 full M-tiles; waves own (matrix, ntile) units, reuse B across
// the 5 M-tiles. K,V kept f32 in LDS. Writes attn probs (d_out) + attn_out (ws)
// ---------------------------------------------------------------------------
__global__ __launch_bounds__(256)
void kv_attn(const float* __restrict__ nbrf, const float* __restrict__ nbrt,
             const float* __restrict__ nbre, const int* __restrict__ masks,
             const _Float16* __restrict__ Bk, const _Float16* __restrict__ Bv,
             const float* __restrict__ Qws, float* __restrict__ AOws,
             float* __restrict__ attnProbs) {
  extern __shared__ char smem[];
  _Float16* KV = (_Float16*)smem;                         // 80*448 f16
  float* Kl = (float*)(smem + (size_t)80 * KVP * 2);      // 80*272 f32
  float* Vl = Kl + 80 * QD;                               // 80*272 f32
  float* qb = Vl + 80 * QD;                               // 4*272
  float* sb = qb + 4 * QD;                                // 160 scores
  float* pb = sb + 160;                                   // 160 probs

  int tid = threadIdx.x;
  int blk = blockIdx.x;

  // build kv_in = [nbr_node | nbr_edge | nbr_time] rows (f16, zero-padded)
  for (int i = tid; i < 80 * KVP; i += 256) {
    int r = i / KVP, c = i % KVP;
    size_t gr = (size_t)blk * 80 + r;   // global (root,neighbor) row
    float v = 0.0f;
    if (c < NODE_D)                 v = nbrf[gr * NODE_D + c];
    else if (c < NODE_D + EDGE_D)   v = nbre[gr * EDGE_D + (c - NODE_D)];
    else if (c < KD)                v = nbrt[gr * TIME_D + (c - NODE_D - EDGE_D)];
    KV[i] = (_Float16)v;
  }
  for (int i = tid; i < 4 * QD; i += 256) {
    int j = i / QD, c = i % QD;
    qb[i] = Qws[((size_t)blk * 4 + j) * QD + c];
  }
  __syncthreads();

  int lane = tid & 31, wv = tid >> 5;
  int arow = lane & 15;
  int aoff = (lane >> 4) * 8;
  for (int u = wv; u < 2 * NTILES; u += 8) {          // 17 units for K, 17 for V
    const _Float16* Bw = (u < NTILES) ? Bk : Bv;
    float* Dst         = (u < NTILES) ? Kl : Vl;
    int nt             = (u < NTILES) ? u : (u - NTILES);
    v8f acc[5] = {};
    for (int kt = 0; kt < KT_KV; kt++) {
      union { uint4 u4[2]; v16h h; } b;
      const uint4* pb2 = (const uint4*)&Bw[((size_t)nt * KT_KV + kt) * 512 + lane * 16];
      b.u4[0] = pb2[0]; b.u4[1] = pb2[1];
      for (int mt = 0; mt < 5; mt++) {
        union { uint4 u4[2]; v16h h; } a;
        const uint4* pa = (const uint4*)&KV[(mt * 16 + arow) * KVP + kt * 32 + (lane >> 4) * 16];
        a.u4[0] = pa[0]; a.u4[1] = pa[1];
        acc[mt] = __builtin_amdgcn_wmma_f32_16x16x32_f16(false, a.h, false, b.h,
                                                         (short)0, acc[mt], false, false);
      }
    }
    int col = nt * 16 + arow;
    for (int mt = 0; mt < 5; mt++)
      for (int r = 0; r < 8; r++)
        Dst[(mt * 16 + aoff + r) * QD + col] = acc[mt][r];
  }
  __syncthreads();

  // scores: one thread per (root j, head h, neighbor n)
  if (tid < 160) {
    int j = tid / 40, h = (tid / 20) % 2, n = tid % 20;
    size_t rb = (size_t)blk * 4 + j;
    const float* qr = &qb[j * QD + h * HEADD];
    const float* kr = &Kl[(j * 20 + n) * QD + h * HEADD];
    float s = 0.0f;
    for (int d = 0; d < HEADD; d++) s = fmaf(qr[d], kr[d], s);
    s *= kScale;
    if (masks[rb * NN + n] == 0) s = -1e10f;
    sb[tid] = s;
  }
  __syncthreads();

  // softmax per (root, head)
  if (tid < 8) {
    int j = tid >> 1, h = tid & 1;
    float* s = &sb[j * 40 + h * 20];
    float mx = s[0];
    for (int n = 1; n < 20; n++) mx = fmaxf(mx, s[n]);
    float sum = 0.0f;
    for (int n = 0; n < 20; n++) { float e = __expf(s[n] - mx); pb[j*40 + h*20 + n] = e; sum += e; }
    float inv = 1.0f / sum;
    size_t rb = (size_t)blk * 4 + j;
    for (int n = 0; n < 20; n++) {
      float p = pb[j*40 + h*20 + n] * inv;
      pb[j*40 + h*20 + n] = p;
      attnProbs[rb * (NH * NN) + h * NN + n] = p;
    }
  }
  __syncthreads();

  // attn_out = probs @ V
  for (int i = tid; i < 4 * QD; i += 256) {
    int j = i / QD, c = i % QD;
    int h = c / HEADD;
    const float* p = &pb[j * 40 + h * 20];
    float o = 0.0f;
    for (int n = 0; n < 20; n++) o = fmaf(p[n], Vl[(j * 20 + n) * QD + c], o);
    AOws[((size_t)blk * 4 + j) * QD + c] = o;
  }
}

// ---------------------------------------------------------------------------
// Kernel 3: proj = attn_out @ Wr^T + br + residual; LayerNorm -> y (16 roots/blk)
// ---------------------------------------------------------------------------
__global__ __launch_bounds__(256)
void out_proj_ln(const float* __restrict__ AOws, const _Float16* __restrict__ Br,
                 const float* __restrict__ nodef, const float* __restrict__ timef,
                 const float* __restrict__ br, const float* __restrict__ gamma,
                 const float* __restrict__ beta, float* __restrict__ Y) {
  __shared__ _Float16 A[16 * QP];
  __shared__ float P[16 * QD];
  int tid = threadIdx.x, blk = blockIdx.x;
  for (int i = tid; i < 16 * QP; i += 256) {
    int r = i / QP, c = i % QP;
    float v = (c < QD) ? AOws[((size_t)blk * 16 + r) * QD + c] : 0.0f;
    A[i] = (_Float16)v;
  }
  __syncthreads();

  int lane = tid & 31, wv = tid >> 5;
  int arow = lane & 15;
  int aoff = (lane >> 4) * 8;
  for (int nt = wv; nt < NTILES; nt += 8) {
    v8f acc = {};
    for (int kt = 0; kt < KT_Q; kt++) {
      union { uint4 u4[2]; v16h h; } a, b;
      const uint4* pa = (const uint4*)&A[arow * QP + kt * 32 + (lane >> 4) * 16];
      a.u4[0] = pa[0]; a.u4[1] = pa[1];
      const uint4* pb = (const uint4*)&Br[((size_t)nt * KT_Q + kt) * 512 + lane * 16];
      b.u4[0] = pb[0]; b.u4[1] = pb[1];
      acc = __builtin_amdgcn_wmma_f32_16x16x32_f16(false, a.h, false, b.h,
                                                   (short)0, acc, false, false);
    }
    int col = nt * 16 + arow;
    for (int r = 0; r < 8; r++) P[(aoff + r) * QD + col] = acc[r];
  }
  __syncthreads();

  // LayerNorm: wave wv handles rows 2*wv, 2*wv+1 (wave32 shuffles)
  for (int m = wv * 2; m < wv * 2 + 2; m++) {
    size_t root = (size_t)blk * 16 + m;
    float xs[9];
    float sum = 0.0f;
    for (int i = 0; i < 9; i++) {
      int c = lane + i * 32;
      float x = 0.0f;
      if (c < QD) {
        float res = (c < NODE_D) ? nodef[root * NODE_D + c]
                                 : timef[root * TIME_D + (c - NODE_D)];
        x = P[m * QD + c] + br[c] + res;
      }
      xs[i] = x; sum += x;
    }
    for (int o = 16; o > 0; o >>= 1) sum += __shfl_xor(sum, o, 32);
    float mu = sum / (float)QD;
    float vs = 0.0f;
    for (int i = 0; i < 9; i++) {
      int c = lane + i * 32;
      if (c < QD) { float d = xs[i] - mu; vs += d * d; }
    }
    for (int o = 16; o > 0; o >>= 1) vs += __shfl_xor(vs, o, 32);
    float inv = rsqrtf(vs / (float)QD + kEps);
    for (int i = 0; i < 9; i++) {
      int c = lane + i * 32;
      if (c < QD) Y[root * QD + c] = (xs[i] - mu) * inv * gamma[c] + beta[c];
    }
  }
}

// ---------------------------------------------------------------------------
extern "C" void kernel_launch(void* const* d_in, const int* in_sizes, int n_in,
                              void* d_out, int out_size, void* d_ws, size_t ws_size,
                              hipStream_t stream) {
  const float* nodef = (const float*)d_in[0];
  const float* nodet = (const float*)d_in[1];
  const float* nbrf  = (const float*)d_in[2];
  const float* nbrt  = (const float*)d_in[3];
  const float* nbre  = (const float*)d_in[4];
  const int*   mask  = (const int*)d_in[5];
  const float* Wq    = (const float*)d_in[6];
  const float* Wk    = (const float*)d_in[7];
  const float* Wv    = (const float*)d_in[8];
  const float* Wr    = (const float*)d_in[9];
  const float* br    = (const float*)d_in[10];
  const float* gamma = (const float*)d_in[11];
  const float* beta  = (const float*)d_in[12];

  char* ws = (char*)d_ws;
  _Float16* Bq  = (_Float16*)(ws + OFF_BQ);
  _Float16* Brr = (_Float16*)(ws + OFF_BR);
  _Float16* Bk  = (_Float16*)(ws + OFF_BK);
  _Float16* Bv  = (_Float16*)(ws + OFF_BV);
  float* Qws  = (float*)(ws + OFF_Q);
  float* AOws = (float*)(ws + OFF_AO);

  float* Yout = (float*)d_out;
  float* Aout = Yout + (size_t)BB * QD;   // attn probs [B, H, N]

  int nq = NTILES * KT_Q  * 512;
  int nk = NTILES * KT_KV * 512;
  swizzle_w<<<(nq + 255) / 256, 256, 0, stream>>>(Wq, Bq,  QD, KT_Q);
  swizzle_w<<<(nq + 255) / 256, 256, 0, stream>>>(Wr, Brr, QD, KT_Q);
  swizzle_w<<<(nk + 255) / 256, 256, 0, stream>>>(Wk, Bk,  KD, KT_KV);
  swizzle_w<<<(nk + 255) / 256, 256, 0, stream>>>(Wv, Bv,  KD, KT_KV);

  q_gemm<<<BB / 16, 256, 0, stream>>>(nodef, nodet, Bq, Qws);

  size_t smem2 = (size_t)80 * KVP * 2      // kv f16
               + (size_t)80 * QD * 4 * 2   // K,V f32
               + (size_t)4 * QD * 4        // q
               + 160 * 4 * 2;              // scores + probs
  kv_attn<<<BB / 4, 256, smem2, stream>>>(nbrf, nbrt, nbre, mask, Bk, Bv,
                                          Qws, AOws, Aout);

  out_proj_ln<<<BB / 16, 256, 0, stream>>>(AOws, Brr, nodef, nodet,
                                           br, gamma, beta, Yout);
}
